// HetGNN_47373489275211
// MI455X (gfx1250) — matrix operand order repaired and checked by reading, not compiled, not run on hardware
//
#include <hip/hip_runtime.h>
#include <hip/hip_bf16.h>
#include <math.h>

typedef __attribute__((ext_vector_type(16))) __bf16 v16bf;
typedef __attribute__((ext_vector_type(8)))  float  v8f;
typedef __attribute__((ext_vector_type(4))) unsigned u32x4;
typedef __attribute__((ext_vector_type(8))) int      i32x8;
typedef __attribute__((ext_vector_type(4))) int      i32x4;

#define N_NODES 20000
#define DIM     128
#define HID     64
#define WIH_ELEMS (256 * 128)   // one direction of Wih, bf16 (64 KB)
#define WHH_ELEMS (256 * 64)    // one direction of Whh, bf16 (32 KB)
#define WPB     2               // waves per block (1250 tiles = 625 blocks x 2)

__device__ __forceinline__ float sigmoidf_(float x) {
  return 1.0f / (1.0f + __expf(-x));
}

__device__ __forceinline__ v8f wmma_bf16(v16bf a, v16bf b, v8f c) {
  // D = A(16x32 bf16) * B(32x16 bf16) + C(16x16 f32)
  return __builtin_amdgcn_wmma_f32_16x16x32_bf16(false, a, false, b, (short)0, c,
                                                 false, false);
}

// B-fragment (32K x 16N) from bf16 weights stored [n][Krow] row-major.
// Works for global or LDS source (addrspace inferred after inlining).
__device__ __forceinline__ v16bf load_bfrag(const __bf16* W, int Krow, int nf,
                                            int kk, int lane) {
  int n  = lane & 15;
  int kb = kk * 32 + ((lane >> 4) & 1) * 8;
  const __bf16* p = W + (nf * 16 + n) * Krow + kb;
  v16bf b;
#pragma unroll
  for (int i = 0; i < 8; ++i) { b[i] = p[i]; b[8 + i] = p[16 + i]; }
  return b;
}

#if __has_builtin(__builtin_amdgcn_tensor_load_to_lds)
// One 1-D TDM copy: `elems` bf16 from global to LDS (elems < 65536).
__device__ __forceinline__ void tdm_copy(__bf16* dstLds, const __bf16* src,
                                         unsigned elems) {
  unsigned ldsoff = (unsigned)(size_t)(void*)dstLds;  // LDS byte offset
  unsigned long long ga = (unsigned long long)(size_t)src;
  u32x4 g0;
  i32x8 g1;
  i32x4 g2 = {0, 0, 0, 0}, g3 = {0, 0, 0, 0};
  i32x8 g4 = {0, 0, 0, 0, 0, 0, 0, 0};
  g0[0] = 1u;                                          // count = 1 (valid D#)
  g0[1] = ldsoff;                                      // lds_addr
  g0[2] = (unsigned)ga;                                // global_addr[31:0]
  g0[3] = (unsigned)((ga >> 32) & 0x1FFFFFFull) | (2u << 30);  // addr hi | type=2
  g1[0] = (int)(1u << 16);                             // data_size = 2 bytes
  g1[1] = (int)((elems & 0xFFFFu) << 16);              // tensor_dim0[15:0]
  g1[2] = (int)((elems >> 16) | (1u << 16));           // dim0 hi | tensor_dim1 = 1
  g1[3] = (int)((elems & 0xFFFFu) << 16);              // tile_dim0 = elems
  g1[4] = 1;                                           // tile_dim1 = 1
  g1[5] = (int)elems;                                  // tensor_dim0_stride lo
  g1[6] = 0;
  g1[7] = 0;
  __builtin_amdgcn_tensor_load_to_lds(g0, g1, g2, g3, g4, 0);
}
#endif

// Stage one direction of Wih (64 KB) + Whh (32 KB) into LDS, block-wide.
__device__ __forceinline__ void stage_weights(__bf16* wihLds, __bf16* whhLds,
                                              const __bf16* wih,
                                              const __bf16* whh) {
#if __has_builtin(__builtin_amdgcn_tensor_load_to_lds)
  if (threadIdx.x < 32) {  // one wave issues both TDM ops (EXEC ignored by TDM)
    tdm_copy(wihLds, wih, WIH_ELEMS);
    tdm_copy(whhLds, whh, WHH_ELEMS);
    __builtin_amdgcn_s_wait_tensorcnt(0);
  }
#else
  for (int i = threadIdx.x; i < WIH_ELEMS; i += blockDim.x) wihLds[i] = wih[i];
  for (int i = threadIdx.x; i < WHH_ELEMS; i += blockDim.x) whhLds[i] = whh[i];
#endif
  __syncthreads();
}

// Batched-16 bidirectional LSTM + mean over time, one wave per 16-node tile;
// waves in the block share the LDS-staged weights, dir loop in lockstep.
// MODE 0: x from f32 contents [16][T][128].  MODE 1: x rows gathered from bf16
// content_h via nbr indices [16][T].
template <int MODE>
__device__ void bilstm16(const float* xf, const __bf16* chs, const int* nbr, int T,
                         const __bf16* WihG,  // [2][256][128] bf16 (global)
                         const __bf16* WhhG,  // [2][256][64]  bf16 (global)
                         const float* bias,   // [2][256] f32
                         float* outf,         // [16][128]
                         __bf16* outbf,       // [16][128] or nullptr
                         __bf16* wihLds,      // LDS 256*128 bf16 (one dir)
                         __bf16* whhLds,      // LDS 256*64  bf16 (one dir)
                         float* hbuf) {       // LDS [16][64], per wave
  const int lane = threadIdx.x & 31;
  const int m    = lane & 15;        // A-frag row / B-frag col
  const int kg   = (lane >> 4) & 1;  // lane group
  const int n    = lane & 15;
  const float invT = 1.0f / (float)T;

  for (int dir = 0; dir < 2; ++dir) {
    stage_weights(wihLds, whhLds, WihG + (size_t)dir * WIH_ELEMS,
                  WhhG + (size_t)dir * WHH_ELEMS);  // block-wide + barrier

    const float* bd = bias + dir * 256;
    v8f c[4], hs[4];
#pragma unroll
    for (int i = 0; i < 4; ++i)
#pragma unroll
      for (int r = 0; r < 8; ++r) { c[i][r] = 0.0f; hs[i][r] = 0.0f; }

    for (int t = 0; t < T; ++t) {
      int tx = dir ? (T - 1 - t) : t;

      // g = bias (broadcast over rows); 16 N-frags cover the 256 gate columns
      v8f g[16];
#pragma unroll
      for (int nf = 0; nf < 16; ++nf) {
        float bv = bd[nf * 16 + n];
#pragma unroll
        for (int r = 0; r < 8; ++r) g[nf][r] = bv;
      }

      // g += x_t @ Wih^T  (K = 128 -> 4 WMMA K-steps, B frags from LDS)
#pragma unroll
      for (int kk = 0; kk < 4; ++kk) {
        int k0 = kk * 32 + kg * 8;  // halves 0..7 -> k0.., 8..15 -> k0+16..
        v16bf a;
        if constexpr (MODE == 0) {
          const float* rp = xf + (m * T + tx) * DIM + k0;
#pragma unroll
          for (int i = 0; i < 8; ++i) a[i] = (__bf16)rp[i];
#pragma unroll
          for (int i = 0; i < 8; ++i) a[8 + i] = (__bf16)rp[16 + i];
        } else {
          int row = nbr[m * T + tx];
          const __bf16* rp = chs + (size_t)row * DIM + k0;
#pragma unroll
          for (int i = 0; i < 8; ++i) a[i] = rp[i];
#pragma unroll
          for (int i = 0; i < 8; ++i) a[8 + i] = rp[16 + i];
        }
#pragma unroll
        for (int nf = 0; nf < 16; ++nf)
          g[nf] = wmma_bf16(a, load_bfrag(wihLds, 128, nf, kk, lane), g[nf]);
      }

      // g += h_{t-1} @ Whh^T  (K = 64 -> 2 K-steps, B frags from LDS)
      if (t > 0) {
        asm volatile("s_wait_dscnt 0" ::: "memory");  // h written to LDS last step
#pragma unroll
        for (int kk = 0; kk < 2; ++kk) {
          int k0 = kk * 32 + kg * 8;
          const float* hp = hbuf + m * HID + k0;
          v16bf a;
#pragma unroll
          for (int i = 0; i < 8; ++i) a[i] = (__bf16)hp[i];
#pragma unroll
          for (int i = 0; i < 8; ++i) a[8 + i] = (__bf16)hp[16 + i];
#pragma unroll
          for (int nf = 0; nf < 16; ++nf)
            g[nf] = wmma_bf16(a, load_bfrag(whhLds, 64, nf, kk, lane), g[nf]);
        }
      }

      // Gate nonlinearity (i,f,g,o as 64-col chunks of the 256)
#pragma unroll
      for (int cf = 0; cf < 4; ++cf) {
#pragma unroll
        for (int r = 0; r < 8; ++r) {
          float iv = sigmoidf_(g[cf][r]);
          float fv = sigmoidf_(g[4 + cf][r]);
          float gv = tanhf(g[8 + cf][r]);
          float ov = sigmoidf_(g[12 + cf][r]);
          float cc = fv * c[cf][r] + iv * gv;
          c[cf][r] = cc;
          float hv = ov * tanhf(cc);
          hs[cf][r] += hv;
          // D-frag layout -> LDS [row][col] for next step's A-frag transpose
          hbuf[(r + kg * 8) * HID + cf * 16 + n] = hv;
        }
      }
    }

    // mean over time; fwd -> cols [0,64), bwd -> cols [64,128)
#pragma unroll
    for (int cf = 0; cf < 4; ++cf) {
#pragma unroll
      for (int r = 0; r < 8; ++r) {
        int mm  = r + kg * 8;
        int col = dir * HID + cf * 16 + n;
        float v = hs[cf][r] * invT;
        outf[mm * DIM + col] = v;
        if (outbf) outbf[mm * DIM + col] = (__bf16)v;
      }
    }
    __syncthreads();  // all waves done with LDS weights before next dir restages
  }
}

// ---- Kernel 1: weights f32 -> bf16 --------------------------------------
__global__ void hetgnn_cvt(const float* Wih, const float* Whh,
                           __bf16* Wihb, __bf16* Whhb) {
  int i = blockIdx.x * blockDim.x + threadIdx.x;
  const int nih = 6 * 2 * 256 * 128;
  const int nhh = 6 * 2 * 256 * 64;
  if (i < nih) Wihb[i] = (__bf16)Wih[i];
  if (i < nhh) Whhb[i] = (__bf16)Whh[i];
}

// ---- Kernel 2: content biLSTM (LSTMs 0..2) ------------------------------
__global__ __launch_bounds__(32 * WPB) void hetgnn_content(
    const float* ca, const float* cp, const float* cv,
    const __bf16* Wihb, const __bf16* Whhb, const float* bias,
    float* chf, __bf16* chb) {
  __shared__ __bf16 wihLds[WIH_ELEMS];
  __shared__ __bf16 whhLds[WHH_ELEMS];
  __shared__ float hbuf[WPB][16 * HID];
  int w    = threadIdx.x >> 5;
  int tile = blockIdx.x * WPB + w;
  int typ  = blockIdx.y;
  const float* x = (typ == 0 ? ca : (typ == 1 ? cp : cv)) +
                   (size_t)tile * 16 * 3 * DIM;
  bilstm16<0>(x, nullptr, nullptr, 3,
              Wihb + (size_t)typ * 2 * WIH_ELEMS,
              Whhb + (size_t)typ * 2 * WHH_ELEMS,
              bias + typ * 2 * 256,
              chf + ((size_t)typ * N_NODES + tile * 16) * DIM,
              chb + ((size_t)typ * N_NODES + tile * 16) * DIM,
              wihLds, whhLds, hbuf[w]);
}

// ---- Kernel 3: neighbor biLSTM (LSTMs 3+s) ------------------------------
struct NbrPtrs { const int* p[9]; };

__global__ __launch_bounds__(32 * WPB) void hetgnn_neigh(
    NbrPtrs nb, const __bf16* chb,
    const __bf16* Wihb, const __bf16* Whhb, const float* bias, float* nhf) {
  __shared__ __bf16 wihLds[WIH_ELEMS];
  __shared__ __bf16 whhLds[WHH_ELEMS];
  __shared__ float hbuf[WPB][16 * HID];
  int w    = threadIdx.x >> 5;
  int tile = blockIdx.x * WPB + w;
  int pair = blockIdx.y;          // d*3+s
  int s = pair % 3;
  int l = 3 + s;                  // neighbor LSTM keyed by SOURCE type
  bilstm16<1>(nullptr, chb + (size_t)s * N_NODES * DIM,
              nb.p[pair] + tile * 16 * 10, 10,
              Wihb + (size_t)l * 2 * WIH_ELEMS,
              Whhb + (size_t)l * 2 * WHH_ELEMS,
              bias + l * 2 * 256,
              nhf + ((size_t)pair * N_NODES + tile * 16) * DIM, nullptr,
              wihLds, whhLds, hbuf[w]);
}

// ---- Kernel 4: attention fusion -----------------------------------------
__global__ __launch_bounds__(256) void hetgnn_atten(
    const float* chf, const float* nhf, const float* aw, const float* ab,
    float* out) {
  int wid  = (blockIdx.x * blockDim.x + threadIdx.x) >> 5;  // one wave per node
  int lane = threadIdx.x & 31;
  int typ  = wid / N_NODES;
  int node = wid % N_NODES;
  if (typ >= 3) return;

  const float* self = chf + ((size_t)typ * N_NODES + node) * DIM;
  const float* w1 = aw + typ * 2 * DIM;  // weights for self half of pair
  const float* w2 = w1 + DIM;            // weights for emb half

  float ef[4][4];
  float base = 0.0f;
  float dk[4] = {0.0f, 0.0f, 0.0f, 0.0f};
#pragma unroll
  for (int j = 0; j < 4; ++j) {
    int f = lane + 32 * j;
    float sv = self[f];
    base += sv * w1[f];
#pragma unroll
    for (int k = 0; k < 4; ++k) {
      float ev = (k < 3)
          ? nhf[(((size_t)typ * 3 + k) * N_NODES + node) * DIM + f]
          : sv;
      ef[k][j] = ev;
      dk[k] += ev * w2[f];
    }
  }
#pragma unroll
  for (int off = 16; off > 0; off >>= 1) {
    base += __shfl_xor(base, off, 32);
#pragma unroll
    for (int k = 0; k < 4; ++k) dk[k] += __shfl_xor(dk[k], off, 32);
  }
  float b = ab[typ];
  float sc[4], mx = -1e30f;
#pragma unroll
  for (int k = 0; k < 4; ++k) {
    float v = base + dk[k] + b;
    v = (v > 0.0f) ? v : 0.01f * v;  // leaky_relu
    sc[k] = v;
    mx = fmaxf(mx, v);
  }
  float den = 0.0f;
#pragma unroll
  for (int k = 0; k < 4; ++k) { sc[k] = __expf(sc[k] - mx); den += sc[k]; }
  float inv = 1.0f / den;
#pragma unroll
  for (int j = 0; j < 4; ++j) {
    float acc = 0.0f;
#pragma unroll
    for (int k = 0; k < 4; ++k) acc += sc[k] * inv * ef[k][j];
    out[((size_t)typ * N_NODES + node) * DIM + lane + 32 * j] = acc;
  }
}

extern "C" void kernel_launch(void* const* d_in, const int* in_sizes, int n_in,
                              void* d_out, int out_size, void* d_ws, size_t ws_size,
                              hipStream_t stream) {
  (void)in_sizes; (void)n_in; (void)out_size; (void)ws_size;
  const float* ca = (const float*)d_in[0];
  const float* cp = (const float*)d_in[1];
  const float* cv = (const float*)d_in[2];
  NbrPtrs nb;
  for (int i = 0; i < 9; ++i) nb.p[i] = (const int*)d_in[3 + i];
  const float* Wih = (const float*)d_in[12];
  const float* Whh = (const float*)d_in[13];
  const float* bia = (const float*)d_in[14];
  const float* aw  = (const float*)d_in[15];
  const float* ab  = (const float*)d_in[16];
  float* out = (float*)d_out;

  // Workspace carve (all offsets 16B aligned)
  char* w = (char*)d_ws;
  __bf16* Wihb = (__bf16*)w; w += (size_t)6 * 2 * 256 * 128 * 2;   // 768 KB
  __bf16* Whhb = (__bf16*)w; w += (size_t)6 * 2 * 256 * 64 * 2;    // 384 KB
  float*  chf  = (float*)w;  w += (size_t)3 * N_NODES * DIM * 4;   // 30.7 MB
  __bf16* chb  = (__bf16*)w; w += (size_t)3 * N_NODES * DIM * 2;   // 15.4 MB
  float*  nhf  = (float*)w;                                        // 92.2 MB

  const int nih = 6 * 2 * 256 * 128;
  hipLaunchKernelGGL(hetgnn_cvt, dim3((nih + 255) / 256), dim3(256), 0, stream,
                     Wih, Whh, Wihb, Whhb);
  hipLaunchKernelGGL(hetgnn_content, dim3(625, 3), dim3(32 * WPB), 0, stream,
                     ca, cp, cv, Wihb, Whhb, bia, chf, chb);
  hipLaunchKernelGGL(hetgnn_neigh, dim3(625, 9), dim3(32 * WPB), 0, stream,
                     nb, chb, Wihb, Whhb, bia, nhf);
  hipLaunchKernelGGL(hetgnn_atten, dim3(3 * N_NODES / 8), dim3(256), 0, stream,
                     chf, nhf, aw, ab, out);
}
